// DMPNN_65558380806593
// MI455X (gfx1250) — matrix operand order repaired and checked by reading, not compiled, not run on hardware
//
#include <hip/hip_runtime.h>
#include <hip/hip_bf16.h>

// ---------------------------------------------------------------------------
// CDNA5 (gfx1250) D-MPNN inference.
//   - GEMMs via v_wmma_f32_16x16x32_bf16 (bf16 inputs, f32 accumulate).
//   - Persistent GEMM blocks: full bf16 weight matrix staged into LDS ONCE
//     per block via the Tensor Data Mover (tensor_load_to_lds, TENSORcnt),
//     B-fragments read from LDS (conflict-free padded stride), double-
//     buffered so ds_load latency overlaps the WMMA chain.
//   - Segment sums via fp32 global atomics.
// ---------------------------------------------------------------------------

typedef __attribute__((ext_vector_type(16))) __bf16 v16bf;
typedef __attribute__((ext_vector_type(8)))  float  v8f;
typedef __attribute__((ext_vector_type(4)))  unsigned int u32x4;
typedef __attribute__((ext_vector_type(8)))  int    i32x8;
typedef __attribute__((ext_vector_type(4)))  int    i32x4;
typedef __attribute__((ext_vector_type(4)))  unsigned short u16x4;

union ABFrag { v16bf v; u32x4 q[2]; };

__device__ __forceinline__ unsigned short f2bf(float f) {
  unsigned int u = __float_as_uint(f);
  u += 0x7FFFu + ((u >> 16) & 1u);   // round-to-nearest-even
  return (unsigned short)(u >> 16);
}

// TDM: flat 1-D copy of `nq` 8-byte elements from global -> LDS offset ldsOff.
__device__ __forceinline__ void tdm_load_1d(const void* gptr, unsigned ldsOff, int nq) {
  unsigned long long ga = (unsigned long long)gptr;
  u32x4 g0;
  g0[0] = 1u;                                   // count=1 (valid), user mode
  g0[1] = ldsOff;                               // lds_addr (bytes)
  g0[2] = (unsigned)(ga & 0xFFFFFFFFu);         // global_addr[31:0]
  g0[3] = (unsigned)((ga >> 32) & 0x1FFFFFFu)   // global_addr[56:32]
          | (2u << 30);                         // type = 2 ("image")
  i32x8 g1;
  g1[0] = 0x00030000;                           // data_size=3 (8B), no pad/iter
  g1[1] = (int)((unsigned)(nq & 0xFFFF) << 16); // tensor_dim0[15:0]
  g1[2] = (int)(((unsigned)nq >> 16) & 0xFFFFu) // tensor_dim0[31:16]
          | (1 << 16);                          // tensor_dim1 = 1
  g1[3] = (int)((unsigned)nq << 16);            // tile_dim0 = nq (<=65535)
  g1[4] = 1;                                    // tile_dim1 = 1
  g1[5] = nq;                                   // tensor_dim0_stride[31:0]
  g1[6] = 0;
  g1[7] = 0;
  i32x4 z4 = {0, 0, 0, 0};
  i32x8 z8 = {0, 0, 0, 0, 0, 0, 0, 0};
  // 6-arg variant (clang-23 / therock-10.0 headers): groups 0..3 + spare + cpol
  __builtin_amdgcn_tensor_load_to_lds(g0, g1, z4, z4, z8, 0);
}

// --------------------------- weight conversion ------------------------------
// out is [304 x KPS] bf16 row-major (KPS = KP+8 padded stride), zero padded.
__global__ __launch_bounds__(256) void cvt_weight(const float* __restrict__ W,
                                                  unsigned short* __restrict__ out,
                                                  int K, int KPS) {
  int idx = blockIdx.x * 256 + threadIdx.x;
  int total = 304 * KPS;
  if (idx >= total) return;
  int r = idx / KPS;
  int k = idx - r * KPS;
  float v = (r < 300 && k < K) ? W[r * K + k] : 0.0f;
  out[idx] = f2bf(v);
}

// ------------------------------- zero ---------------------------------------
__global__ __launch_bounds__(256) void zero_kernel(float* __restrict__ p, unsigned int n) {
  unsigned int i = blockIdx.x * 256u + threadIdx.x;
  if (i < n) p[i] = 0.0f;
}

// --------------------------- segment scatter-add ----------------------------
__global__ __launch_bounds__(256) void scatter_add300(const float* __restrict__ src,
                                                      const int* __restrict__ idx,
                                                      float* __restrict__ dst,
                                                      unsigned int nRows) {
  unsigned int i = blockIdx.x * 256u + threadIdx.x;
  unsigned int total = nRows * 300u;
  if (i >= total) return;
  unsigned int r = i / 300u;
  unsigned int c = i - r * 300u;
  atomicAdd(&dst[(size_t)idx[r] * 300u + c], src[i]);
}

// ------------------------- WMMA gather-GEMM-ReLU ----------------------------
// Persistent blocks: weight [304 x KPS] bf16 staged to LDS once via TDM, then
// loop over 16-row tiles.  out[r,0:300] = relu(A[r,:] @ W^T (+bias)(+resid)).
//
// MODE 0: A = concat(x[src[r]] (133), edge_attr[r] (14)), KP=160
// MODE 1: A = nei[src[r]] (300), KP=320, residual = m (in-place ok)
// MODE 2: A = concat(x[r] (133), atom_msg[r] (300)), KP=448, +bias
// MODE 3: A = mol[r] (300), KP=320, +bias
template<int MODE, int KP>
__global__ __launch_bounds__(256) void gemm_wmma(
    const float* __restrict__ A0, const float* __restrict__ A1,
    const int* __restrict__ gidx,
    const unsigned short* __restrict__ Wbf,
    const float* __restrict__ bias,
    const float* __restrict__ resid,
    float* __restrict__ out, int nTiles) {
  constexpr int KPS    = KP + 8;            // padded stride (conflict-free)
  constexpr int WBYTES = 304 * KPS * 2;     // weight bytes in LDS
  constexpr int NQ     = WBYTES / 8;        // qwords for TDM (<= 65535)
  constexpr int NK     = KP / 32;           // K-steps
  extern __shared__ unsigned char smem[];   // dynamic LDS, base offset 0
  unsigned short* ldsW = (unsigned short*)smem;
  unsigned short* ldsA = (unsigned short*)(smem + WBYTES);

  // ---- one TDM copy of the whole weight matrix into LDS (wave 0 only) ----
  if (threadIdx.x < 32) {
    tdm_load_1d(Wbf, 0u, NQ);
    __builtin_amdgcn_s_wait_tensorcnt(0);
  }

  const int wave = threadIdx.x >> 5;
  const int lane = threadIdx.x & 31;
  const int half = lane >> 4;               // which 16-lane half
  const int nloc = lane & 15;               // A row m / output col-in-tile
  // A fragment: lane holds row m=lane&15; VGPRs 0-3 = K {half*8..+7},
  // VGPRs 4-7 = K {16+half*8..+7}  -> two contiguous 16B LDS chunks.
  const unsigned short* arow = ldsA + nloc * KPS + half * 8;

  for (int tile = blockIdx.x; tile < nTiles; tile += gridDim.x) {
    const int rowBase = tile * 16;

    // ---- stage 16 gathered rows (bf16) into LDS ----
    if (MODE == 1 || MODE == 3) {
      // 300 = 75 float4 chunks per row: vector loads + packed 8B LDS stores
      for (int i = threadIdx.x; i < 16 * 75; i += 256) {
        int r = i / 75;
        int c = (i - r * 75) * 4;
        size_t srow = (MODE == 1) ? (size_t)gidx[rowBase + r] : (size_t)(rowBase + r);
        float4 v4 = *(const float4*)(A0 + srow * 300 + c);
        u16x4 o;
        o[0] = f2bf(v4.x); o[1] = f2bf(v4.y); o[2] = f2bf(v4.z); o[3] = f2bf(v4.w);
        *(u16x4*)(ldsA + r * KPS + c) = o;
      }
      for (int i = threadIdx.x; i < 16 * (KP - 300); i += 256) {  // zero K pad
        int r = i / (KP - 300);
        int k = 300 + (i - r * (KP - 300));
        ldsA[r * KPS + k] = 0;
      }
    } else {
      for (int i = threadIdx.x; i < 16 * KP; i += 256) {
        int r = i / KP;
        int k = i - r * KP;
        int gr = rowBase + r;
        float val = 0.0f;
        if (MODE == 0) {
          if (k < 133)      val = A0[(size_t)gidx[gr] * 133 + k];
          else if (k < 147) val = A1[(size_t)gr * 14 + (k - 133)];
        } else {  // MODE 2
          if (k < 133)      val = A0[(size_t)gr * 133 + k];
          else if (k < 433) val = A1[(size_t)gr * 300 + (k - 133)];
        }
        ldsA[r * KPS + k] = f2bf(val);
      }
    }
    __syncthreads();   // A staged; also orders TDM-filled ldsW on first iter

    for (int nt = wave; nt < 19; nt += 8) {            // 19 n-tiles cover 304
      const int col = nt * 16 + nloc;
      // B fragment: lane holds col n; K {half*8..+7}, {16+half*8..+7} (matches A)
      const unsigned short* wrow = ldsW + col * KPS + half * 8;
      v8f acc = {0.f, 0.f, 0.f, 0.f, 0.f, 0.f, 0.f, 0.f};
      ABFrag bcur, bnxt;
      bcur.q[0] = *(const u32x4*)(wrow);
      bcur.q[1] = *(const u32x4*)(wrow + 16);
      #pragma unroll
      for (int j = 0; j < NK; ++j) {
        const int kb = j * 32;
        if (j + 1 < NK) {                              // double-buffer B
          bnxt.q[0] = *(const u32x4*)(wrow + kb + 32);
          bnxt.q[1] = *(const u32x4*)(wrow + kb + 48);
        }
        ABFrag a;
        a.q[0] = *(const u32x4*)(arow + kb);
        a.q[1] = *(const u32x4*)(arow + kb + 16);
        acc = __builtin_amdgcn_wmma_f32_16x16x32_bf16(
            /*neg_a=*/false, a.v, /*neg_b=*/false, bcur.v,
            /*c_mod=*/(short)0, acc, /*reuse_a=*/false, /*reuse_b=*/false);
        bcur = bnxt;
      }
      if (col < 300) {                                 // drop N padding
        #pragma unroll
        for (int d = 0; d < 8; ++d) {
          int row = rowBase + d + half * 8;            // C/D fragment layout
          float v = acc[d];
          if (MODE == 2 || MODE == 3) v += bias[col];
          if (MODE == 1)              v += resid[(size_t)row * 300 + col];
          v = fmaxf(v, 0.0f);
          out[(size_t)row * 300 + col] = v;
        }
      }
    }
    __syncthreads();   // protect ldsA before next tile restages it
  }
}

// ----------------------------- final 300->1 dot -----------------------------
__global__ __launch_bounds__(256) void ffn2_kernel(const float* __restrict__ h,
                                                   const float* __restrict__ w,
                                                   const float* __restrict__ b,
                                                   float* __restrict__ out) {
  int mol = blockIdx.x * 8 + (threadIdx.x >> 5);       // one wave per molecule
  int lane = threadIdx.x & 31;
  float s = 0.0f;
  for (int c = lane; c < 300; c += 32) s += h[(size_t)mol * 300 + c] * w[c];
  for (int off = 16; off > 0; off >>= 1) s += __shfl_xor(s, off, 32);
  if (lane == 0) out[mol] = s + b[0];
}

// ---------------------------------------------------------------------------
extern "C" void kernel_launch(void* const* d_in, const int* in_sizes, int n_in,
                              void* d_out, int out_size, void* d_ws, size_t ws_size,
                              hipStream_t stream) {
  (void)in_sizes; (void)n_in; (void)out_size; (void)ws_size;
  const float* x     = (const float*)d_in[0];
  const int*   eidx  = (const int*)  d_in[1];   // [2, 400000]
  const float* eattr = (const float*)d_in[2];
  const int*   batch = (const int*)  d_in[3];
  const float* W_i   = (const float*)d_in[4];
  const float* W_h   = (const float*)d_in[5];
  const float* W_o_w = (const float*)d_in[6];
  const float* W_o_b = (const float*)d_in[7];
  const float* f1w   = (const float*)d_in[8];
  const float* f1b   = (const float*)d_in[9];
  const float* f2w   = (const float*)d_in[10];
  const float* f2b   = (const float*)d_in[11];
  float* out = (float*)d_out;

  const int* src = eidx;
  const int* tgt = eidx + 400000;

  char* ws = (char*)d_ws;
  float*          m    = (float*)(ws);                       // 480,000,000 B
  float*          nei  = (float*)(ws + 480000000);           // 120,000,000 B
  float*          mol  = (float*)(ws + 600000000);           //   4,915,200 B
  unsigned short* wiB  = (unsigned short*)(ws + 604915200);  // 304*168*2 = 102,144
  unsigned short* whB  = (unsigned short*)(ws + 605017344);  // 304*328*2 = 199,424
  unsigned short* woB  = (unsigned short*)(ws + 605216768);  // 304*456*2 = 277,248
  unsigned short* f1B  = (unsigned short*)(ws + 605494016);  // 304*328*2 = 199,424
  float*          h    = (float*)(ws + 605693440);           //   4,915,200 B
  float*          atom_h = m;                                // reuse m buffer

  // dynamic LDS: weight (304*KPS*2) + A tile (16*KPS*2)
  const size_t smem0 = (size_t)(304 + 16) * 168 * 2;   // 107,520 B
  const size_t smem1 = (size_t)(304 + 16) * 328 * 2;   // 209,920 B
  const size_t smem2 = (size_t)(304 + 16) * 456 * 2;   // 291,840 B

  // 1) weights -> bf16, padded [304 x (KP+8)]
  cvt_weight<<<(304 * 168 + 255) / 256, 256, 0, stream>>>(W_i,   wiB, 147, 168);
  cvt_weight<<<(304 * 328 + 255) / 256, 256, 0, stream>>>(W_h,   whB, 300, 328);
  cvt_weight<<<(304 * 456 + 255) / 256, 256, 0, stream>>>(W_o_w, woB, 433, 456);
  cvt_weight<<<(304 * 328 + 255) / 256, 256, 0, stream>>>(f1w,   f1B, 300, 328);

  // 2) m = relu(concat(x[src], edge_attr) @ W_i^T)     [400000 x 300]
  gemm_wmma<0, 160><<<1024, 256, smem0, stream>>>(x, eattr, src, wiB,
                                                  nullptr, nullptr, m, 25000);

  // 3) two message-passing rounds: m = relu(m + nei[src] @ W_h^T)
  for (int it = 0; it < 2; ++it) {
    zero_kernel<<<(30000000u + 255) / 256, 256, 0, stream>>>(nei, 30000000u);
    scatter_add300<<<(120000000u + 255) / 256, 256, 0, stream>>>(m, tgt, nei, 400000u);
    gemm_wmma<1, 320><<<1024, 256, smem1, stream>>>(nei, nullptr, src, whB,
                                                    nullptr, m, m, 25000);
  }

  // 4) atom_msg = segment_sum(m, tgt)  -> nei
  zero_kernel<<<(30000000u + 255) / 256, 256, 0, stream>>>(nei, 30000000u);
  scatter_add300<<<(120000000u + 255) / 256, 256, 0, stream>>>(m, tgt, nei, 400000u);

  // 5) atom_h = relu(concat(x, atom_msg) @ W_o^T + b)  [100000 x 300] (into m)
  gemm_wmma<2, 448><<<1024, 256, smem2, stream>>>(x, nei, nullptr, woB,
                                                  W_o_b, nullptr, atom_h, 6250);

  // 6) mol = segment_sum(atom_h, batch)                [4096 x 300]
  zero_kernel<<<(1228800u + 255) / 256, 256, 0, stream>>>(mol, 1228800u);
  scatter_add300<<<(30000000u + 255) / 256, 256, 0, stream>>>(atom_h, batch, mol, 100000u);

  // 7) h = relu(mol @ ffn1^T + b)                      [4096 x 300]
  gemm_wmma<3, 320><<<256, 256, smem1, stream>>>(mol, nullptr, nullptr, f1B,
                                                 f1b, nullptr, h, 256);

  // 8) out = h @ ffn2^T + b                            [4096 x 1]
  ffn2_kernel<<<512, 256, 0, stream>>>(h, f2w, f2b, out);
}
